// attention_unit_30219389895262
// MI455X (gfx1250) — matrix operand order repaired and checked by reading, not compiled, not run on hardware
//
#include <hip/hip_runtime.h>
#include <hip/hip_bf16.h>
#include <cstdint>

#define BB 16
#define CC 256
#define NN 2048
#define CQd 64

typedef __bf16 bf16;
typedef __attribute__((ext_vector_type(16))) __bf16 v16bf;
typedef __attribute__((ext_vector_type(8)))  __bf16 v8bf;
typedef __attribute__((ext_vector_type(8)))  float  v8f;
typedef __attribute__((ext_vector_type(4)))  unsigned int u32x4;
typedef __attribute__((ext_vector_type(8)))  int i32x8;
typedef __attribute__((ext_vector_type(4)))  int i32x4;

static __device__ __forceinline__ v16bf cat16(v8bf lo, v8bf hi) {
  return __builtin_shufflevector(lo, hi, 0,1,2,3,4,5,6,7,8,9,10,11,12,13,14,15);
}
// A-fragment loader: lane holds 16 contiguous K elems [kb, kb+16),
// kb = k0 + (lane>=16 ? 16 : 0) (same k order as the B fragments).
static __device__ __forceinline__ v16bf load_a16(const bf16* row, int kb) {
  return *(const v16bf*)(row + kb);
}
static __device__ __forceinline__ v8f vzero8() {
  v8f z = {0.f,0.f,0.f,0.f,0.f,0.f,0.f,0.f};
  return z;
}
#define WMMA_BF16(A,Bv,Cv) \
  __builtin_amdgcn_wmma_f32_16x16x32_bf16(false,(A),false,(Bv),(short)0,(Cv),false,false)

// ---------------------------------------------------------------------------
// TDM descriptor (ISA ch.8): load a 2D tile (tile_dim0=32 k-elems x
// tile_dim1=256 rows, bf16) from a 2048-stride tensor into LDS, with LDS
// padding of 4 DWORDs after every 16 DWORDs (row pitch 64B -> 80B).
// Toolchain here exposes the 6-arg builtin (clang-23 / therock-10.0 form).
// ---------------------------------------------------------------------------
static __device__ __forceinline__ void tdm_load_h_slab(unsigned lds_off, const void* gptr) {
  unsigned long long ga = (unsigned long long)(uintptr_t)gptr;
  u32x4 g0;
  g0[0] = 1u;                                            // count=1 (valid), user mode
  g0[1] = lds_off;                                       // lds_addr (bytes)
  g0[2] = (unsigned)(ga & 0xFFFFFFFFu);                  // global_addr[31:0]
  g0[3] = (unsigned)((ga >> 32) & 0x1FFFFFFu) | (2u << 30); // addr[56:32] | type=2
  i32x8 g1;
  g1[0] = (1 << 16)        // data_size = 2 bytes
        | (1 << 20)        // pad_enable
        | (3 << 22)        // pad_interval: 16 DWORDs (64B) between pads
        | (3 << 25);       // pad_amount: 4 DWORDs (16B)
  g1[1] = (int)((2048u & 0xFFFFu) << 16);                // tensor_dim0 = 2048 (lo16)
  g1[2] = (int)(256u << 16);                             // dim0 hi16=0 | tensor_dim1=256 lo16
  g1[3] = (int)(32u << 16);                              // dim1 hi16=0 | tile_dim0 = 32
  g1[4] = 256;                                           // tile_dim1 = 256, tile_dim2 = 0
  g1[5] = 2048;                                          // tensor_dim0_stride[31:0]
  g1[6] = 0;                                             // stride hi | dim1_stride lo
  g1[7] = 0;
  i32x4 g2 = {0,0,0,0};                                  // 2D tensor: groups 2/3 unused
  i32x4 g3 = {0,0,0,0};
  i32x8 g4 = {0,0,0,0,0,0,0,0};                          // extra group (6-arg form)
  __builtin_amdgcn_tensor_load_to_lds(g0, g1, g2, g3, g4, 0);
}

// ---------------------------------------------------------------------------
// Kernel 1a: fp32 inputs [B,C,N] -> bf16 xT [B,N,C] (LDS tile transpose)
// ---------------------------------------------------------------------------
__global__ __launch_bounds__(256) void k_transpose_cvt(const float* __restrict__ in,
                                                       __hip_bfloat16* __restrict__ xT) {
  __shared__ float tile[32][33];
  int b = blockIdx.z, c0 = blockIdx.y * 32, n0 = blockIdx.x * 32;
  int t = threadIdx.x;
  int tn = t & 31, tc = t >> 5;
  #pragma unroll
  for (int k = 0; k < 4; ++k) {
    int c = tc + 8 * k;
    tile[c][tn] = in[((size_t)b * CC + c0 + c) * NN + n0 + tn];
  }
  __syncthreads();
  int tn2 = t >> 3, cg = t & 7;
  union { __hip_bfloat16 h[4]; uint2 u; } pk;
  #pragma unroll
  for (int i = 0; i < 4; ++i) pk.h[i] = __float2bfloat16(tile[cg * 4 + i][tn2]);
  *(uint2*)&xT[((size_t)b * NN + n0 + tn2) * CC + c0 + cg * 4] = pk.u;
}

// ---------------------------------------------------------------------------
// Kernel 1b: plain fp32 -> bf16 convert (weights)
// ---------------------------------------------------------------------------
__global__ __launch_bounds__(256) void k_cvt(const float* __restrict__ s,
                                             __hip_bfloat16* __restrict__ d, int n) {
  int i = blockIdx.x * 256 + threadIdx.x;
  if (i < n) d[i] = __float2bfloat16(s[i]);
}

// ---------------------------------------------------------------------------
// Kernel 2: f/g projection.  outT[b][n][cq] = relu(W x + bias), CQ=64, K=C=256.
// A = W rows (M=cq), B = xT rows (N=n).  Wave: 64cq x 16n, 8 k-steps.
// ---------------------------------------------------------------------------
__global__ __launch_bounds__(256) void k_proj_fg(const __hip_bfloat16* __restrict__ Wq,
                                                 const float* __restrict__ bias,
                                                 const __hip_bfloat16* __restrict__ xT,
                                                 __hip_bfloat16* __restrict__ outT) {
  int w = blockIdx.x * 8 + (threadIdx.x >> 5);
  int lane = threadIdx.x & 31, lm = lane & 15, hi = lane >> 4;
  int b = w >> 7, nt = w & 127, n0 = nt * 16;
  const bf16* xrow = (const bf16*)xT + ((size_t)b * NN + n0 + lm) * CC;
  const bf16* Wp = (const bf16*)Wq;
  v8f acc[4];
  #pragma unroll
  for (int j = 0; j < 4; ++j) acc[j] = vzero8();
  for (int k0 = 0; k0 < CC; k0 += 32) {
    v16bf bfrag = *(const v16bf*)(xrow + k0 + hi * 16);
    #pragma unroll
    for (int j = 0; j < 4; ++j) {
      v16bf afrag = load_a16(Wp + (size_t)(16 * j + lm) * CC, k0 + hi * 16);
      acc[j] = WMMA_BF16(afrag, bfrag, acc[j]);
    }
  }
  __hip_bfloat16* orow = outT + ((size_t)b * NN + n0 + lm) * CQd;
  #pragma unroll
  for (int j = 0; j < 4; ++j) {
    union { __hip_bfloat16 h[8]; uint4 u; } pk;
    const float* bp = bias + 16 * j + 8 * hi;
    #pragma unroll
    for (int r = 0; r < 8; ++r)
      pk.h[r] = __float2bfloat16(fmaxf(acc[j][r] + bp[r], 0.f));
    *(uint4*)&orow[16 * j + 8 * hi] = pk.u;
  }
}

// ---------------------------------------------------------------------------
// Kernel 3: h projection, written row-major h[b][c][n].
// D[M=n][N=c]: A = xT rows (M=n), B = Wh rows (col c).  Wave: 16n x 64c.
// ---------------------------------------------------------------------------
__global__ __launch_bounds__(256) void k_proj_h(const __hip_bfloat16* __restrict__ Whq,
                                                const float* __restrict__ bias,
                                                const __hip_bfloat16* __restrict__ xT,
                                                __hip_bfloat16* __restrict__ hout) {
  int w = blockIdx.x * 8 + (threadIdx.x >> 5);
  int lane = threadIdx.x & 31, lm = lane & 15, hi = lane >> 4;
  int b = w >> 9, rem = w & 511, nt = rem >> 2, cg = rem & 3;
  int n0 = nt * 16, c0 = cg * 64;
  const bf16* xrow = (const bf16*)xT + ((size_t)b * NN + n0 + lm) * CC;
  const bf16* Wp = (const bf16*)Whq;
  v8f acc[4];
  #pragma unroll
  for (int j = 0; j < 4; ++j) acc[j] = vzero8();
  for (int k0 = 0; k0 < CC; k0 += 32) {
    v16bf afrag = load_a16(xrow, k0 + hi * 16);
    #pragma unroll
    for (int j = 0; j < 4; ++j) {
      v16bf bfrag = *(const v16bf*)(Wp + (size_t)(c0 + 16 * j + lm) * CC + k0 + hi * 16);
      acc[j] = WMMA_BF16(afrag, bfrag, acc[j]);
    }
  }
  #pragma unroll
  for (int j = 0; j < 4; ++j) {
    int c = c0 + 16 * j + lm;
    float bv = bias[c];
    union { __hip_bfloat16 h[8]; uint4 u; } pk;
    #pragma unroll
    for (int r = 0; r < 8; ++r)
      pk.h[r] = __float2bfloat16(fmaxf(acc[j][r] + bv, 0.f));
    *(uint4*)&hout[((size_t)b * CC + c) * NN + n0 + 8 * hi] = pk.u;
  }
}

// ---------------------------------------------------------------------------
// Kernel 4: scores + softmax (over m) -> betaT[b][m][n] bf16.
// Per block: one batch, one 16-row n-block.  s tile = gT(A) x fT(B), K=64.
// Online softmax fully in registers; score tiles cached in 16 v8f accums so
// no recompute pass is needed.
// ---------------------------------------------------------------------------
__global__ __launch_bounds__(256) void k_score_softmax(const __hip_bfloat16* __restrict__ gT,
                                                       const __hip_bfloat16* __restrict__ fT,
                                                       __hip_bfloat16* __restrict__ betaT) {
  __shared__ float pmax[8][16], psum[8][16];
  __shared__ float rmax[16], rinv[16];
  int wv = threadIdx.x >> 5, lane = threadIdx.x & 31, lm = lane & 15, hi = lane >> 4;
  int b = blockIdx.x >> 7, nt = blockIdx.x & 127, n0 = nt * 16;
  const bf16* grow = (const bf16*)gT + ((size_t)b * NN + n0 + lm) * CQd;
  v16bf a0 = load_a16(grow, hi * 16);
  v16bf a1 = load_a16(grow, 32 + hi * 16);
  float rm[8], rs[8];
  v8f st[16];
  #pragma unroll
  for (int r = 0; r < 8; ++r) { rm[r] = -1e30f; rs[r] = 0.f; }

  #pragma unroll 2
  for (int i = 0; i < 16; ++i) {
    int m0 = (wv + 8 * i) * 16;
    const bf16* frow = (const bf16*)fT + ((size_t)b * NN + m0 + lm) * CQd;
    v16bf b0 = *(const v16bf*)(frow + hi * 16);
    v16bf b1 = *(const v16bf*)(frow + 32 + hi * 16);
    v8f c = vzero8();
    c = WMMA_BF16(a0, b0, c);
    c = WMMA_BF16(a1, b1, c);
    st[i] = c;
    #pragma unroll
    for (int r = 0; r < 8; ++r) {
      float v = c[r];               // row n0+8*hi+r, col m0+lm
      float tm = v;
      tm = fmaxf(tm, __shfl_xor(tm, 1, 32));
      tm = fmaxf(tm, __shfl_xor(tm, 2, 32));
      tm = fmaxf(tm, __shfl_xor(tm, 4, 32));
      tm = fmaxf(tm, __shfl_xor(tm, 8, 32));
      float nm = fmaxf(rm[r], tm);
      float ts = __expf(v - nm);
      ts += __shfl_xor(ts, 1, 32);
      ts += __shfl_xor(ts, 2, 32);
      ts += __shfl_xor(ts, 4, 32);
      ts += __shfl_xor(ts, 8, 32);
      rs[r] = rs[r] * __expf(rm[r] - nm) + ts;
      rm[r] = nm;
    }
  }
  if (lm == 0) {
    #pragma unroll
    for (int r = 0; r < 8; ++r) { pmax[wv][8 * hi + r] = rm[r]; psum[wv][8 * hi + r] = rs[r]; }
  }
  __syncthreads();
  if (threadIdx.x < 16) {
    int t = threadIdx.x;
    float mx = -1e30f;
    for (int w = 0; w < 8; ++w) mx = fmaxf(mx, pmax[w][t]);
    float s = 0.f;
    for (int w = 0; w < 8; ++w) s += psum[w][t] * __expf(pmax[w][t] - mx);
    rmax[t] = mx;
    rinv[t] = 1.f / s;
  }
  __syncthreads();
  float mloc[8], iloc[8];
  #pragma unroll
  for (int r = 0; r < 8; ++r) { mloc[r] = rmax[8 * hi + r]; iloc[r] = rinv[8 * hi + r]; }

  #pragma unroll 2
  for (int i = 0; i < 16; ++i) {
    int m0 = (wv + 8 * i) * 16;
    union { __hip_bfloat16 h[8]; uint4 u; } pk;
    #pragma unroll
    for (int r = 0; r < 8; ++r)
      pk.h[r] = __float2bfloat16(__expf(st[i][r] - mloc[r]) * iloc[r]);
    *(uint4*)&betaT[((size_t)b * NN + m0 + lm) * NN + n0 + 8 * hi] = pk.u;
  }
}

// ---------------------------------------------------------------------------
// Kernel 5: oT[m][c] = sum_n betaT[m][n]*h[c][n];  out = gamma*o + inputs.
// Block = (batch, group of 8 m-tiles); each wave owns one m-tile x ALL 256 c.
// The shared 256x32 bf16 h k-slab is DMA'd into LDS by the Tensor Data Mover
// (double-buffered, TENSORcnt-tracked, 80B padded row pitch for bank spread);
// barriers publish completion to the other 7 waves.
// ---------------------------------------------------------------------------
#define HPITCH 40   // bf16 elems per LDS row: 32 data + 8 pad (64B + 16B)
__global__ __launch_bounds__(256) void k_out(const __hip_bfloat16* __restrict__ betaT,
                                             const __hip_bfloat16* __restrict__ hbuf,
                                             const float* __restrict__ inp,
                                             const float* __restrict__ gamma,
                                             float* __restrict__ out) {
  __shared__ __attribute__((aligned(128))) __hip_bfloat16 hsl[2][CC * HPITCH]; // 2 x 20KB
  int wv = threadIdx.x >> 5, lane = threadIdx.x & 31, lm = lane & 15, hi = lane >> 4;
  int b = blockIdx.x >> 4;            // 16 blocks per batch
  int mg = blockIdx.x & 15;           // m-tile group
  int m0 = (mg * 8 + wv) * 16;
  const bf16* brow = (const bf16*)betaT + ((size_t)b * NN + m0 + lm) * NN;
  const bf16* hbase = (const bf16*)hbuf + (size_t)b * CC * NN;

  v8f acc[16];
  #pragma unroll
  for (int j = 0; j < 16; ++j) acc[j] = vzero8();

  unsigned lds0 = (unsigned)(uintptr_t)(&hsl[0][0]);
  unsigned lds1 = (unsigned)(uintptr_t)(&hsl[1][0]);

  if (wv == 0) tdm_load_h_slab(lds0, hbase);            // k-slab 0
  for (int ks = 0; ks < 64; ++ks) {
    int k0 = ks * 32;
    if (wv == 0) {
      if (ks + 1 < 64) {
        tdm_load_h_slab((ks & 1) ? lds0 : lds1, hbase + k0 + 32);
        __builtin_amdgcn_s_wait_tensorcnt(1);           // slab ks has landed
      } else {
        __builtin_amdgcn_s_wait_tensorcnt(0);
      }
    }
    __syncthreads();                                    // publish slab ks
    const bf16* lb = (const bf16*)&hsl[ks & 1][0];
    v16bf a = load_a16(brow, k0 + hi * 16);
    __builtin_prefetch((const void*)(brow + k0 + 512), 0, 1);
    #pragma unroll
    for (int j = 0; j < 16; ++j) {
      v16bf bfrag = *(const v16bf*)(lb + (16 * j + lm) * HPITCH + hi * 16);
      acc[j] = WMMA_BF16(a, bfrag, acc[j]);
    }
    __syncthreads();                                    // done reading before next issue
  }

  // D: M = m = m0 + 8*hi + r (contiguous), N = c = 16j + lm
  float gm = gamma[0];
  #pragma unroll
  for (int j = 0; j < 16; ++j) {
    int c = 16 * j + lm;
    size_t base = ((size_t)b * CC + c) * NN + m0 + 8 * hi;
    v8f iv = *(const v8f*)(inp + base);
    v8f ov;
    #pragma unroll
    for (int r = 0; r < 8; ++r) ov[r] = gm * acc[j][r] + iv[r];
    *(v8f*)(out + base) = ov;
  }
}

// ---------------------------------------------------------------------------
extern "C" void kernel_launch(void* const* d_in, const int* in_sizes, int n_in,
                              void* d_out, int out_size, void* d_ws, size_t ws_size,
                              hipStream_t stream) {
  (void)in_sizes; (void)n_in; (void)out_size; (void)ws_size;
  const float* inp   = (const float*)d_in[0];
  const float* Wf    = (const float*)d_in[1];
  const float* bfv   = (const float*)d_in[2];
  const float* Wg    = (const float*)d_in[3];
  const float* bgv   = (const float*)d_in[4];
  const float* Wh    = (const float*)d_in[5];
  const float* bhv   = (const float*)d_in[6];
  const float* gamma = (const float*)d_in[7];
  float* out = (float*)d_out;

  char* wsb = (char*)d_ws;
  size_t off = 0;
  auto alloc = [&](size_t bytes) -> char* {
    char* p = wsb + off;
    off += (bytes + 255) & ~(size_t)255;
    return p;
  };
  __hip_bfloat16* xT  = (__hip_bfloat16*)alloc((size_t)BB * NN * CC * 2);   // 16 MB
  __hip_bfloat16* Wfb = (__hip_bfloat16*)alloc((size_t)CQd * CC * 2);
  __hip_bfloat16* Wgb = (__hip_bfloat16*)alloc((size_t)CQd * CC * 2);
  __hip_bfloat16* Whb = (__hip_bfloat16*)alloc((size_t)CC * CC * 2);
  __hip_bfloat16* fT  = (__hip_bfloat16*)alloc((size_t)BB * NN * CQd * 2);  // 4 MB
  __hip_bfloat16* gT  = (__hip_bfloat16*)alloc((size_t)BB * NN * CQd * 2);  // 4 MB
  __hip_bfloat16* hb  = (__hip_bfloat16*)alloc((size_t)BB * CC * NN * 2);   // 16 MB
  __hip_bfloat16* btT = (__hip_bfloat16*)alloc((size_t)BB * NN * NN * 2);   // 128 MB (L2-resident)

  k_transpose_cvt<<<dim3(NN / 32, CC / 32, BB), 256, 0, stream>>>(inp, xT);
  k_cvt<<<dim3((CQd * CC + 255) / 256), 256, 0, stream>>>(Wf, Wfb, CQd * CC);
  k_cvt<<<dim3((CQd * CC + 255) / 256), 256, 0, stream>>>(Wg, Wgb, CQd * CC);
  k_cvt<<<dim3((CC * CC + 255) / 256), 256, 0, stream>>>(Wh, Whb, CC * CC);

  k_proj_fg<<<dim3(BB * (NN / 16) / 8), 256, 0, stream>>>(Wfb, bfv, xT, fT);
  k_proj_fg<<<dim3(BB * (NN / 16) / 8), 256, 0, stream>>>(Wgb, bgv, xT, gT);
  k_proj_h<<<dim3(BB * (NN / 16) * (CC / 64) / 8), 256, 0, stream>>>(Whb, bhv, xT, hb);

  k_score_softmax<<<dim3(BB * (NN / 16)), 256, 0, stream>>>(gT, fT, btT);
  k_out<<<dim3(BB * 16), 256, 0, stream>>>(btT, hb, inp, gamma, out);
}
